// GIN_51170240364966
// MI455X (gfx1250) — compile-verified, matrix operand order checked by reading, not run on hardware
//
#include <hip/hip_runtime.h>
#include <hip/hip_bf16.h>
#include <stdint.h>

// ---- problem constants (match reference) ----
#define DIN 128
#define DH  256
#define CLS 16

typedef __attribute__((ext_vector_type(16))) __bf16 v16bf;
typedef __attribute__((ext_vector_type(8)))  __bf16 v8bf;
typedef __attribute__((ext_vector_type(4)))  __bf16 v4bf;
typedef __attribute__((ext_vector_type(8)))  float  v8f;

// monotonic float <-> uint encoding for atomicMax-based segment_max
__device__ __forceinline__ unsigned f2ord(float f) {
  unsigned u = __float_as_uint(f);
  return (u & 0x80000000u) ? ~u : (u | 0x80000000u);
}
__device__ __forceinline__ float ord2f(unsigned u) {
  unsigned v = (u & 0x80000000u) ? (u & 0x7fffffffu) : ~u;
  return __uint_as_float(v);
}

__global__ void zero_f32_kernel(float* __restrict__ p, long n) {
  long i = (long)blockIdx.x * blockDim.x + threadIdx.x;
  if (i < n) p[i] = 0.0f;
}

__global__ void init_u32_kernel(unsigned* __restrict__ p, unsigned val, int n) {
  int i = blockIdx.x * blockDim.x + threadIdx.x;
  if (i < n) p[i] = val;
}

__global__ void degree_kernel(const int* __restrict__ dst, float* __restrict__ deg, int E) {
  int e = blockIdx.x * blockDim.x + threadIdx.x;
  if (e < E)
    __hip_atomic_fetch_add(deg + dst[e], 1.0f, __ATOMIC_RELAXED, __HIP_MEMORY_SCOPE_AGENT);
}

// scatter-add x[src] into agg[dst]; D floats per row, 4 per thread
template <int D>
__global__ void scatter_add_kernel(const float* __restrict__ x, const int* __restrict__ src,
                                   const int* __restrict__ dst, float* __restrict__ agg, int E) {
  const int CH = D / 4;
  long idx = (long)blockIdx.x * blockDim.x + threadIdx.x;
  if (idx >= (long)E * CH) return;
  int e = (int)(idx / CH);
  int c = (int)(idx % CH) * 4;
  int s = src[e], d = dst[e];
  float4 v = *(const float4*)(x + (size_t)s * D + c);
  float* a = agg + (size_t)d * D + c;
  __hip_atomic_fetch_add(a + 0, v.x, __ATOMIC_RELAXED, __HIP_MEMORY_SCOPE_AGENT);
  __hip_atomic_fetch_add(a + 1, v.y, __ATOMIC_RELAXED, __HIP_MEMORY_SCOPE_AGENT);
  __hip_atomic_fetch_add(a + 2, v.z, __ATOMIC_RELAXED, __HIP_MEMORY_SCOPE_AGENT);
  __hip_atomic_fetch_add(a + 3, v.w, __ATOMIC_RELAXED, __HIP_MEMORY_SCOPE_AGENT);
}

// out_bf16 = bf16( x + agg / max(deg,1) )
template <int D>
__global__ void combine_bf16_kernel(const float* __restrict__ x, const float* __restrict__ agg,
                                    const float* __restrict__ deg, __bf16* __restrict__ out, int N) {
  const int CH = D / 4;
  long idx = (long)blockIdx.x * blockDim.x + threadIdx.x;
  if (idx >= (long)N * CH) return;
  int n = (int)(idx / CH);
  int c = (int)(idx % CH) * 4;
  float dg = deg[n];
  float r = 1.0f / (dg > 1.0f ? dg : 1.0f);
  float4 xv = *(const float4*)(x + (size_t)n * D + c);
  float4 av = *(const float4*)(agg + (size_t)n * D + c);
  v4bf o;
  o.x = (__bf16)(xv.x + av.x * r);
  o.y = (__bf16)(xv.y + av.y * r);
  o.z = (__bf16)(xv.z + av.z * r);
  o.w = (__bf16)(xv.w + av.w * r);
  *(v4bf*)(out + (size_t)n * D + c) = o;
}

// W[k][ncol] f32 -> Wt[ncol][k] bf16  (so B fragments are contiguous per column)
__global__ void convert_wt_kernel(const float* __restrict__ W, __bf16* __restrict__ Wt,
                                  int K, int Ncol) {
  int idx = blockIdx.x * blockDim.x + threadIdx.x;
  if (idx >= K * Ncol) return;
  int k = idx / Ncol, c = idx % Ncol;
  Wt[(size_t)c * K + k] = (__bf16)W[idx];
}

// assemble 16-bit A/B fragment from two contiguous 8-element halves
__device__ __forceinline__ v16bf make_frag(const __bf16* __restrict__ p, int dk, int half) {
  v8bf lo = *(const v8bf*)(p + dk + half * 8);
  v8bf hi = *(const v8bf*)(p + dk + 16 + half * 8);
  v16bf f;
#pragma unroll
  for (int j = 0; j < 8; ++j) { f[j] = lo[j]; f[j + 8] = hi[j]; }
  return f;
}

// Y[nrows, 256] = act( X[nrows,K] @ Wt^T + bias )   via v_wmma_f32_16x16x32_bf16.
// Each wave owns one 16-row stripe and JT=4 column tiles: A loaded once per k-step,
// 4 independent WMMAs issued back-to-back (no RAW between accumulators).
// POOL: fused per-graph segment_max via ordered-uint atomicMax (h2 never stored).
template <int K, bool RELU, bool POOL>
__global__ void gemm_wmma_kernel(const __bf16* __restrict__ X, const __bf16* __restrict__ Wt,
                                 const float* __restrict__ bias, float* __restrict__ Y,
                                 const int* __restrict__ gid, unsigned* __restrict__ pooled,
                                 int nrows) {
  const int JT  = 4;                   // column tiles per wave
  const int NTG = (DH / 16) / JT;      // wave-groups across the 256 columns
  int wave = blockIdx.x * (blockDim.x >> 5) + (threadIdx.x >> 5);
  int lane = threadIdx.x & 31;
  int mt  = wave / NTG;
  int ntg = wave % NTG;
  if (mt * 16 >= nrows) return;        // whole-wave exit: EXEC stays all-ones
  int half = lane >> 4;
  int l15  = lane & 15;
  int arow = mt * 16 + l15; if (arow >= nrows) arow = nrows - 1;
  int ncol0 = ntg * (JT * 16) + l15;

  v8f acc[JT];
#pragma unroll
  for (int j = 0; j < JT; ++j) {
    float bv = bias[ncol0 + j * 16];
#pragma unroll
    for (int r = 0; r < 8; ++r) acc[j][r] = bv;
  }

  const __bf16* Xp = X + (size_t)arow * K;
#pragma unroll
  for (int dk = 0; dk < K; dk += 32) {
    v16bf A = make_frag(Xp, dk, half);
#pragma unroll
    for (int j = 0; j < JT; ++j) {
      const __bf16* Wp = Wt + (size_t)(ncol0 + j * 16) * K;
      v16bf B = make_frag(Wp, dk, half);
      acc[j] = __builtin_amdgcn_wmma_f32_16x16x32_bf16(false, A, false, B, (short)0, acc[j],
                                                       false, false);
    }
  }

  // ---- epilogue: C/D layout has VGPR r -> row M = mt*16 + 8*half + r, col = ncol ----
  const int row0 = mt * 16 + 8 * half;
  const bool full = (mt * 16 + 16) <= nrows;
  if (full) {
#pragma unroll
    for (int j = 0; j < JT; ++j) {
      int ncol = ncol0 + j * 16;
      if (POOL) {
#pragma unroll
        for (int r = 0; r < 8; ++r) {
          float v = acc[j][r];
          int g = gid[row0 + r];
          atomicMax(pooled + (size_t)g * DH + ncol, f2ord(v));
        }
      } else {
        float* Yp = Y + (size_t)row0 * DH + ncol;
#pragma unroll
        for (int r = 0; r < 8; ++r) {
          float v = acc[j][r];
          if (RELU) v = v > 0.0f ? v : 0.0f;
          Yp[(size_t)r * DH] = v;           // immediate-offset stores
        }
      }
    }
  } else {
#pragma unroll
    for (int j = 0; j < JT; ++j) {
      int ncol = ncol0 + j * 16;
#pragma unroll
      for (int r = 0; r < 8; ++r) {
        int row = row0 + r;
        if (row >= nrows) continue;
        float v = acc[j][r];
        if (RELU) v = v > 0.0f ? v : 0.0f;
        if (POOL) {
          int g = gid[row];
          atomicMax(pooled + (size_t)g * DH + ncol, f2ord(v));
        } else {
          Y[(size_t)row * DH + ncol] = v;
        }
      }
    }
  }
}

// out[G, CLS] = where(isfinite(hg), hg, 0) @ Wc + bc     (hg decoded from pooled)
__global__ void head_kernel(const unsigned* __restrict__ pooled, const float* __restrict__ Wc,
                            const float* __restrict__ bc, float* __restrict__ out, int G) {
  int idx = blockIdx.x * blockDim.x + threadIdx.x;
  if (idx >= G * CLS) return;
  int g = idx / CLS, c = idx % CLS;
  float sum = bc[c];
  for (int k = 0; k < DH; ++k) {
    float v = ord2f(pooled[(size_t)g * DH + k]);
    unsigned u = __float_as_uint(v);
    if ((u & 0x7f800000u) == 0x7f800000u) v = 0.0f;   // non-finite (incl. empty graph) -> 0
    sum += v * Wc[(size_t)k * CLS + c];
  }
  out[idx] = sum;
}

static inline size_t align_up(size_t x, size_t a) { return (x + a - 1) & ~(a - 1); }

extern "C" void kernel_launch(void* const* d_in, const int* in_sizes, int n_in,
                              void* d_out, int out_size, void* d_ws, size_t ws_size,
                              hipStream_t stream) {
  const float* in_feat = (const float*)d_in[0];
  const int*   src     = (const int*)  d_in[1];
  const int*   dst     = (const int*)  d_in[2];
  const int*   gid     = (const int*)  d_in[3];
  const float* W1      = (const float*)d_in[4];
  const float* b1      = (const float*)d_in[5];
  const float* W2      = (const float*)d_in[6];
  const float* b2      = (const float*)d_in[7];
  const float* Wc      = (const float*)d_in[8];
  const float* bc      = (const float*)d_in[9];

  const int N = in_sizes[3];
  const int E = in_sizes[1];
  const int G = out_size / CLS;

  // ---- workspace layout ----
  char* ws = (char*)d_ws;
  size_t o = 0;
  float* deg  = (float*)(ws + o); o += (size_t)N * 4;
  float* agg1 = (float*)(ws + o); o += (size_t)N * DIN * 4;
  float* agg2 = (float*)(ws + o); o += (size_t)N * DH * 4;
  const long zero_count = (long)N * (1 + DIN + DH);
  o = align_up(o, 256);
  __bf16* xin  = (__bf16*)(ws + o); o += (size_t)N * DIN * 2;
  o = align_up(o, 256);
  float*  h    = (float*)(ws + o);  o += (size_t)N * DH * 4;
  o = align_up(o, 256);
  __bf16* h2in = (__bf16*)(ws + o); o += (size_t)N * DH * 2;
  o = align_up(o, 256);
  __bf16* W1t  = (__bf16*)(ws + o); o += (size_t)DH * DIN * 2;
  __bf16* W2t  = (__bf16*)(ws + o); o += (size_t)DH * DH * 2;
  o = align_up(o, 256);
  unsigned* pooled = (unsigned*)(ws + o); o += (size_t)G * DH * 4;
  (void)ws_size;

  const int TB = 256;
  const unsigned NEG_INF_ORD = 0x007FFFFFu;     // f2ord(-inf)

  // 0) init
  zero_f32_kernel<<<(unsigned)((zero_count + TB - 1) / TB), TB, 0, stream>>>(deg, zero_count);
  init_u32_kernel<<<(G * DH + TB - 1) / TB, TB, 0, stream>>>(pooled, NEG_INF_ORD, G * DH);

  // 1) degree + layer-1 mean numerator
  degree_kernel<<<(E + TB - 1) / TB, TB, 0, stream>>>(dst, deg, E);
  {
    long t = (long)E * (DIN / 4);
    scatter_add_kernel<DIN><<<(unsigned)((t + TB - 1) / TB), TB, 0, stream>>>(in_feat, src, dst, agg1, E);
  }
  {
    long t = (long)N * (DIN / 4);
    combine_bf16_kernel<DIN><<<(unsigned)((t + TB - 1) / TB), TB, 0, stream>>>(in_feat, agg1, deg, xin, N);
  }

  // weights -> transposed bf16 (once per launch; tiny)
  convert_wt_kernel<<<(DIN * DH + TB - 1) / TB, TB, 0, stream>>>(W1, W1t, DIN, DH);
  convert_wt_kernel<<<(DH * DH + TB - 1) / TB, TB, 0, stream>>>(W2, W2t, DH, DH);

  // 2) GEMM1 + bias + ReLU  (WMMA bf16, f32 accumulate; 4 col-tiles per wave)
  const int mtiles = (N + 15) / 16;
  const long waves = (long)mtiles * ((DH / 16) / 4);
  const unsigned gblocks = (unsigned)((waves + 7) / 8);   // 8 waves / 256-thread block
  gemm_wmma_kernel<DIN, true, false><<<gblocks, TB, 0, stream>>>(
      xin, W1t, b1, h, nullptr, nullptr, N);

  // 3) layer-2 aggregation
  {
    long t = (long)E * (DH / 4);
    scatter_add_kernel<DH><<<(unsigned)((t + TB - 1) / TB), TB, 0, stream>>>(h, src, dst, agg2, E);
  }
  {
    long t = (long)N * (DH / 4);
    combine_bf16_kernel<DH><<<(unsigned)((t + TB - 1) / TB), TB, 0, stream>>>(h, agg2, deg, h2in, N);
  }

  // 4) GEMM2 + bias, fused per-graph segment_max (h2 never materialized)
  gemm_wmma_kernel<DH, false, true><<<gblocks, TB, 0, stream>>>(
      h2in, W2t, b2, nullptr, gid, pooled, N);

  // 5) classifier head
  head_kernel<<<(G * CLS + TB - 1) / TB, TB, 0, stream>>>(pooled, Wc, bc, (float*)d_out, G);
}
